// BigramLanguageModel_68685116998139
// MI455X (gfx1250) — matrix operand order repaired and checked by reading, not compiled, not run on hardware
//
#include <hip/hip_runtime.h>
#include <cmath>

typedef __attribute__((ext_vector_type(16))) __bf16 v16bf;
typedef __attribute__((ext_vector_type(8)))  __bf16 v8bf;
typedef __attribute__((ext_vector_type(8)))  float  v8f;
typedef __attribute__((ext_vector_type(4)))  unsigned int u32x4;
typedef __attribute__((ext_vector_type(8)))  unsigned int u32x8;

namespace {
constexpr int Vv = 32000, Cc = 768, Hh = 12, HSs = 64, FFf = 9216;
constexpr int Bb = 4, Tt = 1024, Mm = Bb * Tt, NB = 3;
constexpr int MODE_QKV = 0, MODE_RES = 1, MODE_RELU_BF16 = 2, MODE_BIASF32 = 3;
}

__device__ __forceinline__ v16bf cat8(v8bf a, v8bf b) {
  return __builtin_shufflevector(a, b, 0, 1, 2, 3, 4, 5, 6, 7, 8, 9, 10, 11, 12, 13, 14, 15);
}

__device__ __forceinline__ v8f wmma_bf16(v16bf a, v16bf b, v8f c) {
  return __builtin_amdgcn_wmma_f32_16x16x32_bf16(false, a, false, b, (short)0, c, false, false);
}

// ---------------------------------------------------------------- embedding
__global__ __launch_bounds__(256) void embed_kernel(const int* __restrict__ src,
                                                    const float* __restrict__ tok,
                                                    const float* __restrict__ pos,
                                                    float* __restrict__ x) {
  int idx = blockIdx.x * 256 + threadIdx.x;
  if (idx >= Mm * Cc) return;
  int bt = idx / Cc, c = idx % Cc;
  int t = bt % Tt;
  x[idx] = tok[(size_t)src[bt] * Cc + c] + pos[(size_t)t * Cc + c];
}

// ---------------------------------------------------------------- layernorm
__global__ __launch_bounds__(256) void ln_kernel(const float* __restrict__ x,
                                                 const float* __restrict__ g,
                                                 const float* __restrict__ b,
                                                 float* __restrict__ out) {
  __shared__ float s1[256], s2[256];
  int row = blockIdx.x, tid = threadIdx.x;
  const float* xr = x + (size_t)row * Cc;
  float a = 0.f, q = 0.f;
  for (int c = tid; c < Cc; c += 256) { float v = xr[c]; a += v; q += v * v; }
  s1[tid] = a; s2[tid] = q;
  __syncthreads();
  for (int st = 128; st > 0; st >>= 1) {
    if (tid < st) { s1[tid] += s1[tid + st]; s2[tid] += s2[tid + st]; }
    __syncthreads();
  }
  float mean = s1[0] * (1.0f / Cc);
  float var = s2[0] * (1.0f / Cc) - mean * mean;
  float inv = rsqrtf(var + 1e-5f);
  float* outr = out + (size_t)row * Cc;
  for (int c = tid; c < Cc; c += 256) outr[c] = (xr[c] - mean) * inv * g[c] + b[c];
}

// ------------------------------------------------ pack wq/wk/wv -> [C, 3C] fp32
__global__ __launch_bounds__(256) void pack_qkv_kernel(const float* __restrict__ wq,
                                                       const float* __restrict__ wk,
                                                       const float* __restrict__ wv,
                                                       float* __restrict__ out) {
  int idx = blockIdx.x * 256 + threadIdx.x;
  if (idx >= Cc * 3 * Cc) return;
  int c = idx / (3 * Cc), n = idx % (3 * Cc);
  int sec = n / Cc, nn = n % Cc;
  int hh = nn >> 6, dd = nn & 63;
  const float* w = (sec == 0) ? wq : ((sec == 1) ? wk : wv);
  out[idx] = w[((size_t)hh * Cc + c) * HSs + dd];  // wq[h][c][d]
}

// ------------------------------------------------------- generic bf16 WMMA GEMM
// C[M,N] = A[M,K] * B[K,N] (+epilogue). 256 threads, 64x128 tile, wave = 16x64.
// LDS tiles staged in bf16 (converted once during coop load); fragments are
// aligned ds_load_b128 pairs matching the ISA 7.12.2 A/B VGPR layouts:
//   A 16-bit frag: lane m, halves = {A[m, hi*8 .. +7], A[m, 16+hi*8 .. +7]}
//   B 16-bit frag: lane n, halves = {B[hi*8 .. +7, n], B[16+hi*8 .. +7, n]}
__global__ __launch_bounds__(256) void gemm_wmma(
    const float* __restrict__ A32, const __bf16* __restrict__ A16, int lda,
    const float* __restrict__ Bw, int ldb,
    const float* __restrict__ bias,
    float* __restrict__ outF, int ldo,
    __bf16* __restrict__ outB, int ldob,
    __bf16* __restrict__ qo, __bf16* __restrict__ ko, __bf16* __restrict__ vo,
    int K, int mode, float qscale) {
  __shared__ alignas(16) __bf16 Alb[64][40];    // A tile bf16, 80B pitch
  __shared__ alignas(16) __bf16 Btb[128][40];   // B^T tile bf16: Btb[n][k]
  int tid = threadIdx.x;
  int wave = tid >> 5, lane = tid & 31;
  int laneN = lane & 15, hi = lane >> 4;
  int m0 = blockIdx.y * 64, n0 = blockIdx.x * 128;
  int mrow = (wave & 3) * 16;     // wave row strip
  int ncol = (wave >> 2) * 64;    // wave col strip

  v8f acc[4];
#pragma unroll
  for (int nt = 0; nt < 4; ++nt)
#pragma unroll
    for (int i = 0; i < 8; ++i) acc[nt][i] = 0.f;

  for (int k0 = 0; k0 < K; k0 += 32) {
    __syncthreads();
    if (A32) {                                   // A: 64x32 fp32 -> bf16, b128 loads
#pragma unroll
      for (int j = 0; j < 2; ++j) {
        int lin4 = tid + 256 * j;                // 512 float4s
        int r = lin4 >> 3, c4 = lin4 & 7;
        float4 av = *(const float4*)&A32[(size_t)(m0 + r) * lda + k0 + c4 * 4];
        __bf16* dst = &Alb[r][c4 * 4];
        dst[0] = (__bf16)av.x; dst[1] = (__bf16)av.y;
        dst[2] = (__bf16)av.z; dst[3] = (__bf16)av.w;
      }
    } else {                                     // A already bf16: straight 16B copy
      int r = tid >> 2, c8 = tid & 3;
      *(uint4*)&Alb[r][c8 * 8] = *(const uint4*)&A16[(size_t)(m0 + r) * lda + k0 + c8 * 8];
    }
#pragma unroll
    for (int j = 0; j < 4; ++j) {                // B: 32x128 fp32, transposed bf16 store
      int lin4 = tid + 256 * j;                  // 1024 float4s
      int kk = lin4 >> 5, nn4 = lin4 & 31;
      float4 bv = *(const float4*)&Bw[(size_t)(k0 + kk) * ldb + n0 + nn4 * 4];
      Btb[nn4 * 4 + 0][kk] = (__bf16)bv.x;
      Btb[nn4 * 4 + 1][kk] = (__bf16)bv.y;
      Btb[nn4 * 4 + 2][kk] = (__bf16)bv.z;
      Btb[nn4 * 4 + 3][kk] = (__bf16)bv.w;
    }
    __syncthreads();

    int arow = mrow + laneN;
    v16bf af = cat8(*(const v8bf*)&Alb[arow][hi * 8],
                    *(const v8bf*)&Alb[arow][16 + hi * 8]);
#pragma unroll
    for (int nt = 0; nt < 4; ++nt) {
      int bcol = ncol + nt * 16 + laneN;
      v16bf bfr = cat8(*(const v8bf*)&Btb[bcol][hi * 8],
                       *(const v8bf*)&Btb[bcol][16 + hi * 8]);
      acc[nt] = wmma_bf16(af, bfr, acc[nt]);
    }
  }

  // epilogue: C/D layout -> element (m = i + 8*hi, n = lane&15) per slot
#pragma unroll
  for (int nt = 0; nt < 4; ++nt) {
#pragma unroll
    for (int i = 0; i < 8; ++i) {
      int m = m0 + mrow + i + 8 * hi;
      int n = n0 + ncol + nt * 16 + laneN;
      float val = acc[nt][i];
      if (mode == MODE_QKV) {
        int sec = n / Cc, nn2 = n % Cc;
        int hh = nn2 >> 6, dd = nn2 & 63;
        int bI = m / Tt, tI = m % Tt;
        size_t idx = (((size_t)bI * Hh + hh) * Tt + tI) * HSs + dd;
        if (sec == 0)      qo[idx] = (__bf16)(val * qscale);   // fold score scale into q
        else if (sec == 1) ko[idx] = (__bf16)val;
        else               vo[idx] = (__bf16)val;
      } else if (mode == MODE_RES) {
        size_t idx = (size_t)m * ldo + n;
        outF[idx] = outF[idx] + val + bias[n];                 // x += proj + b
      } else if (mode == MODE_RELU_BF16) {
        float t2 = val + bias[n];
        outB[(size_t)m * ldob + n] = (__bf16)fmaxf(t2, 0.f);
      } else {  // MODE_BIASF32
        outF[(size_t)m * ldo + n] = val + bias[n];
      }
    }
  }
}

// ------------------------------------------------------- flash attention (causal)
// block = 128 threads (4 waves); block handles (b,h, 64 query rows); wave = 16 rows.
// K tile is DMA'd into LDS by the Tensor Data Mover (wave 0 builds a 2D D# and
// issues tensor_load_to_lds; s_wait_tensorcnt 0 + workgroup barrier publishes it).
__global__ __launch_bounds__(128) void attn_kernel(const __bf16* __restrict__ q,
                                                   const __bf16* __restrict__ k,
                                                   const __bf16* __restrict__ v,
                                                   float* __restrict__ o) {
  __shared__ alignas(16) __bf16 k_lds[32][72];      // K tile [s][d]; 144B pitch (TDM-padded)
  __shared__ alignas(16) __bf16 vT_lds[64][40];     // V tile transposed [d][s]
  __shared__ alignas(16) __bf16 P_lds[4][16][40];   // per-wave P (C/D -> A layout bounce)

  const int nQ = Tt / 64;
  int bid = blockIdx.x;
  int qb = bid % nQ;
  int h = (bid / nQ) % Hh;
  int b = bid / (nQ * Hh);
  int tid = threadIdx.x;
  int wave = tid >> 5, lane = tid & 31;
  int laneN = lane & 15, hi = lane >> 4;
  int qbase = qb * 64 + wave * 16;

  const __bf16* qp = q + ((size_t)(b * Hh + h) * Tt) * HSs;
  const __bf16* kp = k + ((size_t)(b * Hh + h) * Tt) * HSs;
  const __bf16* vp = v + ((size_t)(b * Hh + h) * Tt) * HSs;

  // q A-fragments for K chunks d0 = 0, 32 (score scale already folded into q)
  v16bf qa[2];
#pragma unroll
  for (int c2 = 0; c2 < 2; ++c2) {
    const __bf16* qrow = &qp[(size_t)(qbase + laneN) * HSs + c2 * 32];
    qa[c2] = cat8(*(const v8bf*)&qrow[hi * 8], *(const v8bf*)&qrow[16 + hi * 8]);
  }

  v8f Oacc[4];
#pragma unroll
  for (int nt = 0; nt < 4; ++nt)
#pragma unroll
    for (int i = 0; i < 8; ++i) Oacc[nt][i] = 0.f;
  float Mrow[8], Lrow[8];
#pragma unroll
  for (int i = 0; i < 8; ++i) { Mrow[i] = -1e30f; Lrow[i] = 0.f; }

  int nIter = (qb * 64 + 64) / 32;  // causal extent for this block (uniform)
  for (int it = 0; it < nIter; ++it) {
    int s0 = it * 32;
    __syncthreads();

    if (wave == 0) {
      // Tensor DMA: 32 rows x 64 bf16 from (kp + s0*64) into k_lds with LDS padding
      // (pad_interval: 32 DWORDs = 128B row; pad_amount: 4 DWORDs -> 144B pitch).
      unsigned long long ga = (unsigned long long)(uintptr_t)(kp + (size_t)s0 * HSs);
      unsigned int lds_off = (unsigned int)(uintptr_t)&k_lds[0][0];
      u32x4 g0;
      g0[0] = 1u;                                            // count=1 (valid user D#)
      g0[1] = lds_off;                                       // lds_addr
      g0[2] = (unsigned int)ga;                              // global_addr[31:0]
      g0[3] = (unsigned int)((ga >> 32) & 0x1FFFFFFu) | (2u << 30);  // addr[56:32] | type=2
      u32x8 g1;
      g1[0] = (1u << 16) | (1u << 20) | (4u << 22) | (3u << 25); // data_size=2B,pad_en,int=32DW,amt=4DW
      g1[1] = (64u << 16);                                   // tensor_dim0 = 64 (lo16)
      g1[2] = ((unsigned int)Tt << 16);                      // dim0 hi=0 | tensor_dim1 lo16
      g1[3] = (64u << 16);                                   // dim1 hi=0 | tile_dim0 = 64
      g1[4] = 32u;                                           // tile_dim1 = 32, tile_dim2 = 0
      g1[5] = 64u;                                           // tensor_dim0_stride = 64
      g1[6] = 0u;
      g1[7] = 0u;
      asm volatile("tensor_load_to_lds %0, %1\n\t"
                   "s_wait_tensorcnt 0x0"
                   :: "s"(g0), "s"(g1) : "memory");
    }
    // V tile: b128 global loads, transposed bf16 store into LDS
#pragma unroll
    for (int j = 0; j < 2; ++j) {
      int lin8 = tid + 128 * j;                 // 256 chunks of 8 halves
      int s = lin8 >> 3, d8 = lin8 & 7;
      v8bf vv = *(const v8bf*)&vp[(size_t)(s0 + s) * HSs + d8 * 8];
#pragma unroll
      for (int q2 = 0; q2 < 8; ++q2) vT_lds[d8 * 8 + q2][s] = vv[q2];
    }
    __syncthreads();

    // S = q @ k^T : two 16x16 tiles (n = key index), K = 64 in 2 chunks
    float sv[2][8];
#pragma unroll
    for (int sub = 0; sub < 2; ++sub) {
      v8f accS;
#pragma unroll
      for (int i = 0; i < 8; ++i) accS[i] = 0.f;
      int sl = sub * 16 + laneN;
#pragma unroll
      for (int c2 = 0; c2 < 2; ++c2) {
        const __bf16* krow = &k_lds[sl][c2 * 32];
        v16bf kb = cat8(*(const v8bf*)&krow[hi * 8], *(const v8bf*)&krow[16 + hi * 8]);
        accS = wmma_bf16(qa[c2], kb, accS);
      }
#pragma unroll
      for (int i = 0; i < 8; ++i) sv[sub][i] = accS[i];
    }

    // online softmax: per-row stats replicated across each 16-lane half
#pragma unroll
    for (int i = 0; i < 8; ++i) {
      int tg = qbase + i + 8 * hi;
      float tmax = -1e30f;
#pragma unroll
      for (int sub = 0; sub < 2; ++sub) {
        int sg = s0 + sub * 16 + laneN;
        float val = (sg <= tg) ? sv[sub][i] : -1e30f;
        sv[sub][i] = val;
        tmax = fmaxf(tmax, val);
      }
#pragma unroll
      for (int off = 8; off >= 1; off >>= 1)
        tmax = fmaxf(tmax, __shfl_xor(tmax, off, 16));
      float mn = fmaxf(Mrow[i], tmax);
      float alpha = __expf(Mrow[i] - mn);
      Mrow[i] = mn;
      float ps = 0.f;
#pragma unroll
      for (int sub = 0; sub < 2; ++sub) {
        int sg = s0 + sub * 16 + laneN;
        float p = (sg <= tg) ? __expf(sv[sub][i] - mn) : 0.f;
        P_lds[wave][i + 8 * hi][sub * 16 + laneN] = (__bf16)p;
        ps += p;
      }
#pragma unroll
      for (int off = 8; off >= 1; off >>= 1)
        ps += __shfl_xor(ps, off, 16);
      Lrow[i] = Lrow[i] * alpha + ps;
#pragma unroll
      for (int nt = 0; nt < 4; ++nt) Oacc[nt][i] *= alpha;
    }

    // wave-private LDS bounce: C/D layout -> A layout (same-wave LDS is in-order)
    asm volatile("s_wait_dscnt 0" ::: "memory");
    const __bf16* prow = &P_lds[wave][laneN][0];
    v16bf pa = cat8(*(const v8bf*)&prow[hi * 8], *(const v8bf*)&prow[16 + hi * 8]);

    // O += P @ V (K = 32 keys), 4 column tiles of 16
#pragma unroll
    for (int nt = 0; nt < 4; ++nt) {
      const __bf16* vrow = &vT_lds[nt * 16 + laneN][0];
      v16bf vb = cat8(*(const v8bf*)&vrow[hi * 8], *(const v8bf*)&vrow[16 + hi * 8]);
      Oacc[nt] = wmma_bf16(pa, vb, Oacc[nt]);
    }
  }

  // normalize and write o[b, t, h*64 + d] fp32 (heads concatenated)
#pragma unroll
  for (int i = 0; i < 8; ++i) {
    float invl = 1.0f / Lrow[i];
    int tg = qbase + i + 8 * hi;
#pragma unroll
    for (int nt = 0; nt < 4; ++nt) {
      int cc = h * HSs + nt * 16 + laneN;
      o[((size_t)(b * Tt + tg)) * Cc + cc] = Oacc[nt][i] * invl;
    }
  }
}

// ---------------------------------------------------------------- loss
__global__ void zero_loss_kernel(float* loss) {
  if (threadIdx.x == 0 && blockIdx.x == 0) *loss = 0.f;
}

// single-pass online log-sum-exp + NLL
__global__ __launch_bounds__(256) void ce_kernel(const float* __restrict__ logits,
                                                 const int* __restrict__ tgt,
                                                 float* __restrict__ loss) {
  __shared__ float rm[256], rs[256];
  int row = blockIdx.x, tid = threadIdx.x;
  const float* lr = logits + (size_t)row * Vv;
  float m = -3.4e38f, s = 0.f;
  for (int i = tid; i < Vv; i += 256) {
    float l = lr[i];
    float nm = fmaxf(m, l);
    s = s * __expf(m - nm) + __expf(l - nm);
    m = nm;
  }
  rm[tid] = m; rs[tid] = s;
  __syncthreads();
  for (int st = 128; st > 0; st >>= 1) {
    if (tid < st) {
      float m2 = rm[tid + st], s2 = rs[tid + st];
      float nm = fmaxf(rm[tid], m2);
      rs[tid] = rs[tid] * __expf(rm[tid] - nm) + s2 * __expf(m2 - nm);
      rm[tid] = nm;
    }
    __syncthreads();
  }
  if (tid == 0) {
    float lse = rm[0] + __logf(rs[0]);
    float lt = lr[tgt[row]];
    atomicAdd(loss, (lse - lt) * (1.0f / (float)Mm));
  }
}

// ---------------------------------------------------------------- launcher
extern "C" void kernel_launch(void* const* d_in, const int* in_sizes, int n_in,
                              void* d_out, int out_size, void* d_ws, size_t ws_size,
                              hipStream_t stream) {
  (void)in_sizes; (void)n_in; (void)out_size; (void)ws_size;
  const float* tok   = (const float*)d_in[0];
  const float* pos   = (const float*)d_in[1];
  const float* wq    = (const float*)d_in[2];
  const float* wk    = (const float*)d_in[3];
  const float* wv    = (const float*)d_in[4];
  const float* wproj = (const float*)d_in[5];
  const float* bproj = (const float*)d_in[6];
  const float* g1    = (const float*)d_in[7];
  const float* b1    = (const float*)d_in[8];
  const float* g2    = (const float*)d_in[9];
  const float* b2    = (const float*)d_in[10];
  const float* wf1   = (const float*)d_in[11];
  const float* bf1   = (const float*)d_in[12];
  const float* wf2   = (const float*)d_in[13];
  const float* bf2   = (const float*)d_in[14];
  const float* gf    = (const float*)d_in[15];
  const float* bfv   = (const float*)d_in[16];
  const float* wlm   = (const float*)d_in[17];
  const float* blm   = (const float*)d_in[18];
  const int*   src   = (const int*)d_in[19];
  const int*   tgt   = (const int*)d_in[20];

  float* logits = (float*)d_out;
  float* loss = logits + (size_t)Mm * Vv;

  uint8_t* wsb = (uint8_t*)d_ws;
  size_t off = 0;
  auto take = [&](size_t bytes) -> void* {
    void* p = wsb + off;
    off = (off + bytes + 255) & ~(size_t)255;
    return p;
  };
  float*  x     = (float*)take((size_t)Mm * Cc * 4);
  float*  hbuf  = (float*)take((size_t)Mm * Cc * 4);
  float*  obuf  = (float*)take((size_t)Mm * Cc * 4);
  __bf16* qbuf  = (__bf16*)take((size_t)Mm * Cc * 2);
  __bf16* kbuf  = (__bf16*)take((size_t)Mm * Cc * 2);
  __bf16* vbuf  = (__bf16*)take((size_t)Mm * Cc * 2);
  __bf16* ffa   = (__bf16*)take((size_t)Mm * FFf * 2);
  float*  wpack = (float*)take((size_t)Cc * 3 * Cc * 4);

  float qscale = 1.0f / sqrtf((float)Cc);  // reference scales by n_embd^-0.5

  embed_kernel<<<(Mm * Cc + 255) / 256, 256, 0, stream>>>(src, tok, pos, x);

  for (int i = 0; i < NB; ++i) {
    ln_kernel<<<Mm, 256, 0, stream>>>(x, g1 + i * Cc, b1 + i * Cc, hbuf);
    pack_qkv_kernel<<<(Cc * 3 * Cc + 255) / 256, 256, 0, stream>>>(
        wq + (size_t)i * Hh * Cc * HSs, wk + (size_t)i * Hh * Cc * HSs,
        wv + (size_t)i * Hh * Cc * HSs, wpack);
    gemm_wmma<<<dim3(3 * Cc / 128, Mm / 64), 256, 0, stream>>>(
        hbuf, nullptr, Cc, wpack, 3 * Cc, nullptr,
        nullptr, 0, nullptr, 0, qbuf, kbuf, vbuf, Cc, MODE_QKV, qscale);
    attn_kernel<<<Bb * Hh * (Tt / 64), 128, 0, stream>>>(qbuf, kbuf, vbuf, obuf);
    gemm_wmma<<<dim3(Cc / 128, Mm / 64), 256, 0, stream>>>(
        obuf, nullptr, Cc, wproj + (size_t)i * Cc * Cc, Cc, bproj + i * Cc,
        x, Cc, nullptr, 0, nullptr, nullptr, nullptr, Cc, MODE_RES, 0.f);
    ln_kernel<<<Mm, 256, 0, stream>>>(x, g2 + i * Cc, b2 + i * Cc, hbuf);
    gemm_wmma<<<dim3(FFf / 128, Mm / 64), 256, 0, stream>>>(
        hbuf, nullptr, Cc, wf1 + (size_t)i * Cc * FFf, FFf, bf1 + i * FFf,
        nullptr, 0, ffa, FFf, nullptr, nullptr, nullptr, Cc, MODE_RELU_BF16, 0.f);
    gemm_wmma<<<dim3(Cc / 128, Mm / 64), 256, 0, stream>>>(
        nullptr, ffa, FFf, wf2 + (size_t)i * FFf * Cc, Cc, bf2 + i * Cc,
        x, Cc, nullptr, 0, nullptr, nullptr, nullptr, FFf, MODE_RES, 0.f);
  }

  ln_kernel<<<Mm, 256, 0, stream>>>(x, gf, bfv, hbuf);
  gemm_wmma<<<dim3(Vv / 128, Mm / 64), 256, 0, stream>>>(
      hbuf, nullptr, Cc, wlm, Vv, blm,
      logits, Vv, nullptr, 0, nullptr, nullptr, nullptr, Cc, MODE_BIASF32, 0.f);

  zero_loss_kernel<<<1, 1, 0, stream>>>(loss);
  ce_kernel<<<Mm, 256, 0, stream>>>(logits, tgt, loss);
}
